// RecurrentNetContinual_4733053960302
// MI455X (gfx1250) — matrix operand order repaired and checked by reading, not compiled, not run on hardware
//
#include <hip/hip_runtime.h>
#include <hip/hip_bf16.h>

typedef __bf16 bf16;
typedef __attribute__((ext_vector_type(16))) __bf16 v16bf;
typedef __attribute__((ext_vector_type(8)))  float  v8f;

#define T_LEN 1024
#define NB    128
#define NHID  512
#define NIN   128
#define NOUT  128
#define ALPHA 0.1f

#define HID_OFF   0
#define OUT_OFF   (NB * T_LEN * NHID)                  // 67108864
#define HFIN_OFF  (OUT_OFF + NB * T_LEN * NOUT)        // 83886080

// workspace layout (float element offsets)
#define WS_WEFF   0                      // float[512*128]
#define WS_BEFF   (512 * 128)            // float[512]
#define WS_FLAGS  (WS_BEFF + 512)        // int[32] (16 used)
#define WS_XBUF   (WS_FLAGS + 32)        // bf16[8][2][2][16*256] (262144 B)
#define XHALF     (16 * 256)             // bf16 elements per half-state

union AFrag { v16bf v; uint4 q[2]; };

__device__ __forceinline__ v16bf load_b_frag_f32(const float* __restrict__ p) {
    float t[16];
    *(float4*)&t[0]  = ((const float4*)p)[0];
    *(float4*)&t[4]  = ((const float4*)p)[1];
    *(float4*)&t[8]  = ((const float4*)p)[4];
    *(float4*)&t[12] = ((const float4*)p)[5];
    v16bf v;
#pragma unroll
    for (int e = 0; e < 16; ++e) v[e] = (bf16)t[e];
    return v;
}

__device__ __forceinline__ v8f wmma_bf16(v16bf a, v16bf b, v8f c) {
    return __builtin_amdgcn_wmma_f32_16x16x32_bf16(false, a, false, b,
                                                   (short)0, c, false, false);
}

// ---------------------------------------------------------------------------
// Phase 1: W_eff = W_ih @ W_in, b_eff = W_ih@b_in + b_ih + b_hh, zero flags.
// ---------------------------------------------------------------------------
__global__ __launch_bounds__(256) void rnn_prep_kernel(
    const float* __restrict__ W_in, const float* __restrict__ b_in,
    const float* __restrict__ W_ih, const float* __restrict__ b_ih,
    const float* __restrict__ b_hh,
    float* __restrict__ W_eff, float* __restrict__ b_eff,
    int* __restrict__ flags) {
    int idx = blockIdx.x * blockDim.x + threadIdx.x;   // 0..65535
    if (idx < 32) flags[idx] = 0;                      // reset sync flags
    int n = idx >> 7;       // 0..511
    int i = idx & 127;      // 0..127
    float s = 0.f;
    for (int j = 0; j < NHID; ++j)
        s = fmaf(W_ih[n * NHID + j], W_in[j * NIN + i], s);
    W_eff[n * NIN + i] = s;
    if (i == 0) {
        float b = 0.f;
        for (int j = 0; j < NHID; ++j)
            b = fmaf(W_ih[n * NHID + j], b_in[j], b);
        b_eff[n] = b + b_ih[n] + b_hh[n];
    }
}

// ---------------------------------------------------------------------------
// Phase 2: persistent recurrence. 16 blocks x 512 threads (16 waves).
// Block = (pair, side): pair p owns batch rows [16p,16p+16); side s owns
// hidden columns [256s, 256s+256). Wave w owns the 16-column tile
// ncol0 = 256s + 16w with its full-K W_hh^T slice in registers (128 VGPRs,
// 4 waves/SIMD -> no spill). W_eff^T slice lives in LDS. The two sides
// exchange bf16 half-states per step through a double-buffered ws slot with
// a release/acquire flag handshake (signal-then-spin; deadlock-free).
// ---------------------------------------------------------------------------
__global__ __launch_bounds__(512) void rnn_recurrent_kernel(
    const float* __restrict__ u,      // [NB, T, NIN]
    const float* __restrict__ h0,     // [NB, NHID]
    const float* __restrict__ W_hh,   // [NHID, NHID]
    const float* __restrict__ W_eff,  // [NHID, NIN]
    const float* __restrict__ b_eff,  // [NHID]
    bf16* __restrict__ xbuf,          // [8][2][2][16*256]
    int*  __restrict__ flags,         // [16]
    float* __restrict__ hid_out,      // [NB, T, NHID]
    float* __restrict__ hfin_out) {   // [NB, NHID]
    __shared__ __align__(16) bf16 Hs[16 * NHID];   // 16 KB full hidden state
    __shared__ __align__(16) bf16 Us[16 * NIN];    //  4 KB u_t tile
    __shared__ __align__(16) bf16 We[256 * NIN];   // 64 KB W_eff^T slice

    const int tid  = threadIdx.x;
    const int wave = tid >> 5;           // 0..15
    const int lane = tid & 31;
    const int row  = lane & 15;
    const int hi   = lane >> 4;
    const int pair = blockIdx.x >> 1;    // 0..7
    const int side = blockIdx.x & 1;     // 0/1
    const int b0   = pair * 16;
    const int ncl  = wave * 16;          // local column base within side
    const int nc0  = side * 256 + ncl;   // global column base

    // Persistent W_hh^T B fragments (K=512, 16 chunks, 128 VGPRs).
    v16bf Bh[16];
#pragma unroll
    for (int kc = 0; kc < 16; ++kc)
        Bh[kc] = load_b_frag_f32(W_hh + (size_t)(nc0 + row) * NHID + kc * 32 + hi * 8);
    const float bias = b_eff[nc0 + row];

    // Stage this side's W_eff^T slice into LDS (bf16 row-major [256][128]).
    for (int idx = tid; idx < 256 * NIN; idx += 512)
        We[idx] = (bf16)W_eff[(size_t)(side * 256 + (idx >> 7)) * NIN + (idx & 127)];

    // Full initial hidden state into LDS, own tile into fp32 registers.
    for (int idx = tid; idx < 16 * NHID; idx += 512)
        Hs[idx] = (bf16)h0[(size_t)(b0 + (idx >> 9)) * NHID + (idx & 511)];
    float h[8];
#pragma unroll
    for (int i = 0; i < 8; ++i)
        h[i] = h0[(size_t)(b0 + i + hi * 8) * NHID + nc0 + row];
    __syncthreads();

    const int self_f  = pair * 2 + side;
    const int other_f = pair * 2 + (1 - side);

    for (int t = 0; t < T_LEN; ++t) {
        // Stage u_t tile [16, NIN] fp32 -> bf16 LDS (4 elements/thread).
        {
            int e = tid * 4;
            int m = e >> 7;
            int k = e & (NIN - 1);
            float4 f = *(const float4*)(u + ((size_t)(b0 + m) * T_LEN + t) * NIN + k);
            Us[m * NIN + k]     = (bf16)f.x;
            Us[m * NIN + k + 1] = (bf16)f.y;
            Us[m * NIN + k + 2] = (bf16)f.z;
            Us[m * NIN + k + 3] = (bf16)f.w;
        }
        __syncthreads();

        v8f c;
#pragma unroll
        for (int i = 0; i < 8; ++i) c[i] = bias;

        // u_t @ W_eff^T (K=128), B fragments streamed from LDS.
#pragma unroll
        for (int kc = 0; kc < 4; ++kc) {
            AFrag a, b;
            a.q[0] = *(const uint4*)&Us[row * NIN + kc * 32 + hi * 8];
            a.q[1] = *(const uint4*)&Us[row * NIN + kc * 32 + 16 + hi * 8];
            b.q[0] = *(const uint4*)&We[(ncl + row) * NIN + kc * 32 + hi * 8];
            b.q[1] = *(const uint4*)&We[(ncl + row) * NIN + kc * 32 + 16 + hi * 8];
            c = wmma_bf16(a.v, b.v, c);
        }
        // h @ W_hh^T (K=512), B register-resident.
#pragma unroll
        for (int kc = 0; kc < 16; ++kc) {
            AFrag a;
            a.q[0] = *(const uint4*)&Hs[row * NHID + kc * 32 + hi * 8];
            a.q[1] = *(const uint4*)&Hs[row * NHID + kc * 32 + 16 + hi * 8];
            c = wmma_bf16(a.v, Bh[kc], c);
        }
        __syncthreads();  // all Hs/Us reads done before overwrite

        // ReLU + leaky integrate (fp32); publish own half; stream fp32 out.
        bf16* xw = xbuf + (size_t)(self_f * 2 + (t & 1)) * XHALF;
#pragma unroll
        for (int i = 0; i < 8; ++i) {
            float hn = c[i] > 0.f ? c[i] : 0.f;
            h[i] = (1.0f - ALPHA) * h[i] + ALPHA * hn;
            int m = i + hi * 8;
            bf16 hb = (bf16)h[i];
            Hs[m * NHID + nc0 + row] = hb;
            xw[m * 256 + ncl + row]  = hb;
            hid_out[((size_t)(b0 + m) * T_LEN + t) * NHID + nc0 + row] = h[i];
        }

        if (t + 1 < T_LEN) {
            __threadfence();            // make xbuf writes agent-visible
            __syncthreads();
            if (tid == 0) {
                __hip_atomic_fetch_add(&flags[self_f], 1, __ATOMIC_RELEASE,
                                       __HIP_MEMORY_SCOPE_AGENT);
                while (__hip_atomic_load(&flags[other_f], __ATOMIC_ACQUIRE,
                                         __HIP_MEMORY_SCOPE_AGENT) < t + 1)
                    __builtin_amdgcn_s_sleep(1);
            }
            __syncthreads();
            __threadfence();            // invalidate before reading partner data
            // Pull partner half (8 KB) into Hs: 16 B per thread.
            const bf16* xr = xbuf + (size_t)(other_f * 2 + (t & 1)) * XHALF;
            {
                int base = tid * 8;               // 512 threads * 8 bf16 = 4096
                int m = base >> 8;
                int nl = base & 255;
                uint4 v = *(const uint4*)&xr[m * 256 + nl];
                *(uint4*)&Hs[m * NHID + (1 - side) * 256 + nl] = v;
            }
            __syncthreads();
        }
    }

#pragma unroll
    for (int i = 0; i < 8; ++i)
        hfin_out[(size_t)(b0 + i + hi * 8) * NHID + nc0 + row] = h[i];
}

// ---------------------------------------------------------------------------
// Phase 3: output_list = hidden_list @ W_out^T + b_out.  M = NB*T = 131072.
// 256 blocks x 256 threads (2 waves/SIMD -> W_out fragments stay resident).
// ---------------------------------------------------------------------------
__global__ __launch_bounds__(256) void rnn_output_kernel(
    const float* __restrict__ hid,    // [NB*T, NHID]
    const float* __restrict__ W_out,  // [NOUT, NHID]
    const float* __restrict__ b_out,  // [NOUT]
    float* __restrict__ out) {        // [NB*T, NOUT]
    const int tid  = threadIdx.x;
    const int wave = tid >> 5;
    const int lane = tid & 31;
    const int row  = lane & 15;
    const int hi   = lane >> 4;
    const int wg   = blockIdx.x * 8 + wave;   // 0..2047
    const int n0   = (wg & 7) * 16;
    const int mb   = wg >> 3;                 // 0..255

    v16bf Bw[16];
#pragma unroll
    for (int kc = 0; kc < 16; ++kc)
        Bw[kc] = load_b_frag_f32(W_out + (size_t)(n0 + row) * NHID + kc * 32 + hi * 8);
    const float bias = b_out[n0 + row];

    for (int mt = mb; mt < (NB * T_LEN) / 16; mt += 256) {
        v8f c;
#pragma unroll
        for (int i = 0; i < 8; ++i) c[i] = bias;
#pragma unroll
        for (int kc = 0; kc < 16; ++kc) {
            const float* ap = hid + (size_t)(mt * 16 + row) * NHID + kc * 32 + hi * 8;
            float t[16];
            *(float4*)&t[0]  = ((const float4*)ap)[0];
            *(float4*)&t[4]  = ((const float4*)(ap + 4))[0];
            *(float4*)&t[8]  = ((const float4*)(ap + 16))[0];
            *(float4*)&t[12] = ((const float4*)(ap + 20))[0];
            AFrag a;
#pragma unroll
            for (int e = 0; e < 16; ++e) a.v[e] = (bf16)t[e];
            c = wmma_bf16(a.v, Bw[kc], c);
        }
#pragma unroll
        for (int i = 0; i < 8; ++i)
            out[(size_t)(mt * 16 + i + hi * 8) * NOUT + n0 + row] = c[i];
    }
}

// ---------------------------------------------------------------------------
extern "C" void kernel_launch(void* const* d_in, const int* in_sizes, int n_in,
                              void* d_out, int out_size, void* d_ws, size_t ws_size,
                              hipStream_t stream) {
    const float* u     = (const float*)d_in[0];
    const float* h0    = (const float*)d_in[1];
    const float* W_in  = (const float*)d_in[2];
    const float* b_in  = (const float*)d_in[3];
    const float* W_ih  = (const float*)d_in[4];
    const float* b_ih  = (const float*)d_in[5];
    const float* W_hh  = (const float*)d_in[6];
    const float* b_hh  = (const float*)d_in[7];
    const float* W_out = (const float*)d_in[8];
    const float* b_out = (const float*)d_in[9];

    float* out   = (float*)d_out;
    float* wsf   = (float*)d_ws;
    float* W_eff = wsf + WS_WEFF;
    float* b_eff = wsf + WS_BEFF;
    int*   flags = (int*)(wsf + WS_FLAGS);
    bf16*  xbuf  = (bf16*)(wsf + WS_XBUF);

    rnn_prep_kernel<<<256, 256, 0, stream>>>(W_in, b_in, W_ih, b_ih, b_hh,
                                             W_eff, b_eff, flags);
    rnn_recurrent_kernel<<<16, 512, 0, stream>>>(
        u, h0, W_hh, W_eff, b_eff, xbuf, flags, out + HID_OFF, out + HFIN_OFF);
    rnn_output_kernel<<<256, 256, 0, stream>>>(out + HID_OFF, W_out, b_out,
                                               out + OUT_OFF);
}
